// DQNet_54013508715236
// MI455X (gfx1250) — matrix operand order repaired and checked by reading, not compiled, not run on hardware
//
#include <hip/hip_runtime.h>
#include <math.h>

// ---------------------------------------------------------------------------
// Problem constants (from reference): B=32, N=1024, HD=64, NA=512
// ---------------------------------------------------------------------------
#define BB   32
#define NN   1024
#define HD   64
#define NA   512
#define BN   (BB * NN)        // 32768
#define BNA  (BB * NA)        // 16384

typedef __attribute__((ext_vector_type(16))) __bf16 v16bf;
typedef __attribute__((ext_vector_type(8)))  float  v8f;
typedef __attribute__((ext_vector_type(4)))  unsigned int u32x4;
typedef __attribute__((ext_vector_type(8)))  int          i32x8;
typedef __attribute__((ext_vector_type(4)))  int          i32x4;

#if __has_builtin(__builtin_amdgcn_tensor_load_to_lds) && \
    __has_builtin(__builtin_amdgcn_s_wait_tensorcnt)
#define USE_TDM 1
#else
#define USE_TDM 0
#endif

struct FragB {
    union {
        unsigned int u[8];
        v16bf        v;
    };
};

__device__ __forceinline__ unsigned short f2bf(float f) {
    unsigned int u = __float_as_uint(f);
    unsigned int r = u + 0x7FFFu + ((u >> 16) & 1u);   // round-to-nearest-even
    return (unsigned short)(r >> 16);
}

__device__ __forceinline__ v8f zero8() {
    v8f z = {0.f, 0.f, 0.f, 0.f, 0.f, 0.f, 0.f, 0.f};
    return z;
}

// 16-bit A-matrix fragment (16x32 per WMMA K-chunk), per ISA layout:
//   lane<16 : M=lane,    K = {kc+2v, kc+2v+1} (v<4) / {kc+16+2v,...} (v>=4)
//   lane>=16: same M,    K offsets +8
__device__ __forceinline__ void frag_a(FragB& a, const unsigned int* rowp,
                                       int kc, int half) {
    const unsigned int* p = rowp + ((kc + half * 8) >> 1);
#pragma unroll
    for (int v = 0; v < 4; ++v) {
        a.u[v]     = p[v];       // K = kc + half*8 + 2v
        a.u[4 + v] = p[8 + v];   // K = kc + 16 + half*8 + 2v
    }
}

// 16-bit B-matrix fragment (32x16). B kept transposed in LDS as [n][k]
// (i.e. W row-major), each lane reads contiguous K pairs of its column.
__device__ __forceinline__ void frag_b(FragB& b, const unsigned int* colp,
                                       int kc, int half) {
    const unsigned int* p = colp + ((kc + half * 8) >> 1);
#pragma unroll
    for (int v = 0; v < 4; ++v) {
        b.u[v]     = p[v];
        b.u[4 + v] = p[8 + v];
    }
}

__device__ __forceinline__ v8f wmma_bf16(const FragB& a, const FragB& b, v8f c) {
    return __builtin_amdgcn_wmma_f32_16x16x32_bf16(
        /*neg_a=*/false, a.v, /*neg_b=*/false, b.v,
        /*c_mod=*/(short)0, c, /*reuse_a=*/false, /*reuse_b=*/false);
}

#if USE_TDM
// ---------------------------------------------------------------------------
// TDM descriptor: 2-D bf16 tile (tile_dim0=32 contiguous K elements,
// tile_dim1=64 rows with tensor_dim0_stride=1024), DMA'd Global->LDS with
// hardware padding of 4 DWORDs every 16 DWORDs, so each 64B row lands on an
// 80B (40-ushort) LDS stride == our bank-conflict-free layout.
// Field packing per CDNA5 ISA.
// This toolchain exposes the 6-arg builtin:
//   (u32x4 g0, i32x8 g1, i32x4 g2, i32x4 g3, i32x8 extra, i32 cpol)
// ---------------------------------------------------------------------------
__device__ __forceinline__ void tdm_load_hT(unsigned lds_addr,
                                            unsigned long long gaddr) {
    u32x4 g0;
    g0.x = 0x1u;                                                  // count=1, user mode
    g0.y = lds_addr;                                              // lds_addr [63:32]
    g0.z = (unsigned)(gaddr & 0xFFFFFFFFull);                     // global_addr [95:64]
    g0.w = (unsigned)((gaddr >> 32) & 0x1FFFFFFull) | (2u << 30); // addr hi | type=2
    i32x8 g1;
    g1[0] = (1 << 16)      // data_size = 1 -> 2 bytes
          | (1 << 20)      // pad_enable
          | (3 << 22)      // pad_interval = 16 DWORDs
          | (3 << 25);     // pad_amount   = 4 DWORDs
    g1[1] = (int)(1024u << 16);   // tensor_dim0 = 1024  (bits [79:48] low half)
    g1[2] = (int)(64u << 16);     // tensor_dim1 = 64    (bits [111:80] low half)
    g1[3] = (int)(32u << 16);     // tile_dim0 = 32      (bits [127:112])
    g1[4] = 64;                   // tile_dim1 = 64      (bits [143:128])
    g1[5] = 1024;                 // tensor_dim0_stride  (bits [207:160] low 32)
    g1[6] = 0;
    g1[7] = 0;
    i32x4 z4 = {0, 0, 0, 0};                      // groups 2/3 unused (2-D tensor)
    i32x8 z8 = {0, 0, 0, 0, 0, 0, 0, 0};
    __builtin_amdgcn_tensor_load_to_lds(g0, g1, z4, z4, z8, 0);
}
#endif

// ---------------------------------------------------------------------------
// K0: xl1 = x@W1^T + b1 ; h1 = relu(xl1 + b2)   (GNN step 1: h==0 -> n1==0)
// Emits row-major bf16 h (RNN A-operand) and transposed bf16 hT (TDM tiles).
// ---------------------------------------------------------------------------
__global__ void init_kernel(const float* __restrict__ x,
                            const float* __restrict__ W1,
                            const float* __restrict__ b1,
                            const float* __restrict__ b2,
                            float* __restrict__ xl1,
                            float* __restrict__ h,
                            unsigned short* __restrict__ hbf,
                            unsigned short* __restrict__ hbfT) {
    int idx = blockIdx.x * 256 + threadIdx.x;    // exactly BN*HD threads
    int i = idx >> 6, j = idx & 63;
    float v = x[i * 2 + 0] * W1[j * 2 + 0] +
              x[i * 2 + 1] * W1[j * 2 + 1] + b1[j];
    xl1[idx] = v;
    float hv = v + b2[j];
    hv = hv > 0.f ? hv : 0.f;
    h[idx]   = hv;
    unsigned short bf = f2bf(hv);
    hbf[idx] = bf;
    hbfT[(((i >> 10) << 6) + j) * NN + (i & 1023)] = bf;
}

// ---------------------------------------------------------------------------
// K1: n1 = adj @ h   (batched 1024x1024 x 1024x64), bf16 WMMA, f32 accum.
// One WG = one batch x 64 output rows. 8 waves, each owns two 16x16 tiles.
// adj stays fp32 in memory (fits in 192MB L2 for the 2nd step); converted to
// bf16 while staging into LDS. The h tile is DMA'd by the Tensor Data Mover
// (wave 0 issues TENSOR_LOAD_TO_LDS, overlapping the adj conversion, then
// drains TENSORcnt before the workgroup barrier publishes the tile).
// ---------------------------------------------------------------------------
__global__ void gnn_adj_kernel(const float* __restrict__ adj,
                               const unsigned short* __restrict__ hbfT,
                               unsigned short* __restrict__ n1bf) {
    __shared__ unsigned int sA[64 * 20];   // bf16 [row 0..63][k 0..31], stride 40 ush
    __shared__ unsigned int sB[64 * 20];   // bf16 [col j 0..63][k 0..31], stride 40 ush

    const int batch = blockIdx.x >> 4;
    const int row0  = (blockIdx.x & 15) * 64;
    const int tid   = threadIdx.x;
    const int lane  = tid & 31;
    const int wave  = tid >> 5;
    const int mt    = wave >> 1;         // 0..3 : 16-row group within the 64
    const int nt0   = (wave & 1) * 2;    // 0 or 2 : first of two 16-col groups
    const int half  = lane >> 4;
    const int l15   = lane & 15;

    const unsigned int adj_base = ((unsigned int)batch << 20); // batch*N*N

#if USE_TDM
    // flat LDS addresses have the LDS byte offset in addr[31:0]
    const unsigned sB_lds = (unsigned)(unsigned long long)(void*)sB;
    const unsigned long long hT_base =
        (unsigned long long)(const void*)hbfT +
        ((unsigned long long)batch * (unsigned long long)(HD * NN)) * 2ull;
#endif

    v8f acc0 = zero8();
    v8f acc1 = zero8();

    for (int kc = 0; kc < NN; kc += 32) {
#if USE_TDM
        // async DMA of the 64x32 bf16 h tile into sB (wave 0 drives the TDM)
        if (wave == 0) tdm_load_hT(sB_lds, hT_base + (unsigned long long)kc * 2ull);
#endif
        // stage adj tile 64x32 f32 -> bf16 into LDS (all waves, overlaps TDM)
#pragma unroll
        for (int it = 0; it < 8; ++it) {
            int idx = tid + it * 256;                 // 0..2047
            int r = idx >> 5, c = idx & 31;
            float v = adj[adj_base + (unsigned int)(row0 + r) * NN + kc + c];
            ((unsigned short*)sA)[r * 40 + c] = f2bf(v);
        }
#if !USE_TDM
        // fallback: manual copy of transposed h tile
#pragma unroll
        for (int it = 0; it < 8; ++it) {
            int idx = tid + it * 256;                 // 0..2047
            int j = idx >> 5, k = idx & 31;
            ((unsigned short*)sB)[j * 40 + k] =
                hbfT[((batch << 6) + j) * NN + kc + k];
        }
#endif
        // prefetch next adj chunk (lowers to global_prefetch_b8)
        if (kc + 32 < NN) {
            int r = tid >> 5, c = tid & 31;
            __builtin_prefetch(
                &adj[adj_base + (unsigned int)(row0 + r) * NN + kc + 32 + c], 0, 1);
        }
#if USE_TDM
        if (wave == 0) __builtin_amdgcn_s_wait_tensorcnt(0);
#endif
        __syncthreads();

        FragB a, b0, b1;
        frag_a(a,  sA + (mt * 16 + l15) * 20, 0, half);
        frag_b(b0, sB + (nt0 * 16 + l15) * 20, 0, half);
        frag_b(b1, sB + ((nt0 + 1) * 16 + l15) * 20, 0, half);
        acc0 = wmma_bf16(a, b0, acc0);
        acc1 = wmma_bf16(a, b1, acc1);
        __syncthreads();
    }

    // scatter C/D tiles (lane<16: M=r, N=lane ; lane>=16: M=r+8, N=lane-16)
#pragma unroll
    for (int r = 0; r < 8; ++r) {
        int row = row0 + mt * 16 + r + half * 8;
        unsigned int o = (((unsigned int)(batch << 10) + row) << 6);
        n1bf[o + nt0 * 16 + l15]       = f2bf(acc0[r]);
        n1bf[o + (nt0 + 1) * 16 + l15] = f2bf(acc1[r]);
    }
}

// ---------------------------------------------------------------------------
// K2: h = relu(xl1 + n1 @ W2^T + b2) ; emits f32 h + bf16 row-major +
// bf16 transposed copies.
// One WG = 128 rows; wave w owns rows [w*16, w*16+16) x all 64 cols (4 tiles).
// ---------------------------------------------------------------------------
__global__ void gnn_fc_kernel(const unsigned short* __restrict__ n1bf,
                              const float* __restrict__ W2,
                              const float* __restrict__ b2,
                              const float* __restrict__ xl1,
                              float* __restrict__ h_out,
                              unsigned short* __restrict__ hbf_out,
                              unsigned short* __restrict__ hbfT_out) {
    __shared__ unsigned int sW[64 * 36];   // bf16 W2 [j][k], stride 72 ush

    const int tid  = threadIdx.x;
    const int lane = tid & 31;
    const int wave = tid >> 5;
    const int half = lane >> 4;
    const int l15  = lane & 15;

#pragma unroll
    for (int it = 0; it < 16; ++it) {
        int idx = tid + it * 256;            // 4096 = 64*64
        int j = idx >> 6, k = idx & 63;
        ((unsigned short*)sW)[j * 72 + k] = f2bf(W2[idx]);
    }
    __syncthreads();

    const int r0 = blockIdx.x * 128 + wave * 16;
    v8f acc[4] = {zero8(), zero8(), zero8(), zero8()};

    const unsigned int* arow =
        (const unsigned int*)n1bf + (unsigned int)(r0 + l15) * 32;

#pragma unroll
    for (int kc = 0; kc < HD; kc += 32) {
        FragB a;
        frag_a(a, arow, kc, half);
#pragma unroll
        for (int nt = 0; nt < 4; ++nt) {
            FragB b;
            frag_b(b, sW + (nt * 16 + l15) * 36, kc, half);
            acc[nt] = wmma_bf16(a, b, acc[nt]);
        }
    }

#pragma unroll
    for (int nt = 0; nt < 4; ++nt) {
        int col = nt * 16 + l15;
        float bb = b2[col];
#pragma unroll
        for (int r = 0; r < 8; ++r) {
            int row = r0 + r + half * 8;
            float v = acc[nt][r] + xl1[(row << 6) + col] + bb;
            v = v > 0.f ? v : 0.f;
            h_out[(row << 6) + col] = v;
            unsigned short bf = f2bf(v);
            hbf_out[(row << 6) + col] = bf;
            hbfT_out[(((row >> 10) << 6) + col) * NN + (row & 1023)] = bf;
        }
    }
}

// ---------------------------------------------------------------------------
// K3: sequential 2-layer tanh RNN over t=0..1023 (single WG, 8 waves = the
// eight 16x16 tiles of the 32x64 state). Weights + c0/c1 live in LDS as bf16
// (~55KB of the 320KB WGP LDS). Finishes with ge = c1@Wt6^T + bt6.
// ---------------------------------------------------------------------------
__global__ void rnn_kernel(const unsigned short* __restrict__ hbf,
                           const float* __restrict__ Wih0, const float* __restrict__ bih0,
                           const float* __restrict__ Whh0, const float* __restrict__ bhh0,
                           const float* __restrict__ Wih1, const float* __restrict__ bih1,
                           const float* __restrict__ Whh1, const float* __restrict__ bhh1,
                           const float* __restrict__ Wt6,  const float* __restrict__ bt6,
                           float* __restrict__ ge) {
    __shared__ unsigned int sWih0[64 * 36], sWhh0[64 * 36];
    __shared__ unsigned int sWih1[64 * 36], sWhh1[64 * 36];
    __shared__ unsigned int sWt6[64 * 36];
    __shared__ unsigned int sC0[32 * 36], sC1[32 * 36];  // bf16 [batch][k], stride 72
    __shared__ float sB0[64], sB1[64], sBt6[64];

    const int tid  = threadIdx.x;
    const int lane = tid & 31;
    const int wave = tid >> 5;
    const int half = lane >> 4;
    const int l15  = lane & 15;
    const int mt   = wave >> 2;    // 0..1 : batch 16-row group
    const int nt   = wave & 3;     // 0..3 : hd 16-col group
    const int mrow = mt * 16 + l15;        // this lane's A row (batch idx)
    const int col  = nt * 16 + l15;        // this lane's output column

#pragma unroll
    for (int it = 0; it < 16; ++it) {
        int idx = tid + it * 256;
        int j = idx >> 6, k = idx & 63;
        ((unsigned short*)sWih0)[j * 72 + k] = f2bf(Wih0[idx]);
        ((unsigned short*)sWhh0)[j * 72 + k] = f2bf(Whh0[idx]);
        ((unsigned short*)sWih1)[j * 72 + k] = f2bf(Wih1[idx]);
        ((unsigned short*)sWhh1)[j * 72 + k] = f2bf(Whh1[idx]);
        ((unsigned short*)sWt6)[j * 72 + k]  = f2bf(Wt6[idx]);
    }
    if (tid < 64) {
        sB0[tid]  = bih0[tid] + bhh0[tid];
        sB1[tid]  = bih1[tid] + bhh1[tid];
        sBt6[tid] = bt6[tid];
    }
    for (int it = tid; it < 32 * 36; it += 256) { sC0[it] = 0u; sC1[it] = 0u; }
    __syncthreads();

    const unsigned int* c0row = sC0 + mrow * 36;
    const unsigned int* c1row = sC1 + mrow * 36;

    for (int t = 0; t < NN; ++t) {
        // ---- layer 0: c0 = tanh(xt@Wih0^T + c0@Whh0^T + bias) ----
        v8f acc = zero8();
        const unsigned int* arow =
            (const unsigned int*)hbf + (unsigned int)((mrow << 10) + t) * 32;
        FragB a, b;
#pragma unroll
        for (int kc = 0; kc < HD; kc += 32) {
            frag_a(a, arow, kc, half);
            frag_b(b, sWih0 + col * 36, kc, half);
            acc = wmma_bf16(a, b, acc);
            frag_a(a, c0row, kc, half);
            frag_b(b, sWhh0 + col * 36, kc, half);
            acc = wmma_bf16(a, b, acc);
        }
        __syncthreads();                         // all reads of old c0 done
        float nb0 = sB0[col];
#pragma unroll
        for (int r = 0; r < 8; ++r) {
            int row = mt * 16 + r + half * 8;    // batch idx
            ((unsigned short*)sC0)[row * 72 + col] = f2bf(tanhf(acc[r] + nb0));
        }
        __syncthreads();                         // new c0 visible

        // ---- layer 1: c1 = tanh(c0@Wih1^T + c1@Whh1^T + bias) ----
        v8f acc1 = zero8();
#pragma unroll
        for (int kc = 0; kc < HD; kc += 32) {
            frag_a(a, c0row, kc, half);
            frag_b(b, sWih1 + col * 36, kc, half);
            acc1 = wmma_bf16(a, b, acc1);
            frag_a(a, c1row, kc, half);
            frag_b(b, sWhh1 + col * 36, kc, half);
            acc1 = wmma_bf16(a, b, acc1);
        }
        __syncthreads();                         // all reads of old c1 done
        float nb1 = sB1[col];
#pragma unroll
        for (int r = 0; r < 8; ++r) {
            int row = mt * 16 + r + half * 8;
            ((unsigned short*)sC1)[row * 72 + col] = f2bf(tanhf(acc1[r] + nb1));
        }
        __syncthreads();                         // new c1 visible
    }

    // ---- ge = c1 @ Wt6^T + bt6 ----
    v8f acc = zero8();
    FragB a, b;
#pragma unroll
    for (int kc = 0; kc < HD; kc += 32) {
        frag_a(a, c1row, kc, half);
        frag_b(b, sWt6 + col * 36, kc, half);
        acc = wmma_bf16(a, b, acc);
    }
    float bb = sBt6[col];
#pragma unroll
    for (int r = 0; r < 8; ++r) {
        int row = mt * 16 + r + half * 8;
        ge[(row << 6) + col] = acc[r] + bb;
    }
}

// ---------------------------------------------------------------------------
// K4: R + Q_sa. One thread per action row (16384). Tiny K=8 FC kept on VALU.
// ---------------------------------------------------------------------------
__global__ void action_kernel(const int* __restrict__ label,
                              const int* __restrict__ actions,
                              const float* __restrict__ x,
                              const float* __restrict__ ge,
                              const float* __restrict__ WL12h,
                              const float* __restrict__ bL12h,
                              const float* __restrict__ Wt5,
                              const float* __restrict__ bt5,
                              float* __restrict__ R_out,
                              float* __restrict__ Q_out) {
    int row = blockIdx.x * 256 + threadIdx.x;    // exactly BNA threads
    int rb  = row >> 9;                          // NA = 512 actions per batch
    int ai0 = actions[row * 2 + 0];
    int ai1 = actions[row * 2 + 1];
    const int* lab = label + (rb << 10);
    int a0 = lab[ai0];
    int a1 = lab[ai1];
    int li = lab[(ai0 + NN - 1) & (NN - 1)];     // roll(+1): label[(j-1)%n]
    int ri = lab[(ai1 + 1) & (NN - 1)];          // roll(-1): label[(j+1)%n]
    int base = rb << 10;

    int idx4[4] = {a0 + base, a1 + base, li + base, ri + base};
    float Y[8];
#pragma unroll
    for (int p = 0; p < 4; ++p) {
        Y[p * 2 + 0] = x[idx4[p] * 2 + 0];
        Y[p * 2 + 1] = x[idx4[p] * 2 + 1];
    }
    auto dist = [&](int p, int q) {
        float dx = Y[p * 2] - Y[q * 2];
        float dy = Y[p * 2 + 1] - Y[q * 2 + 1];
        return sqrtf(dx * dx + dy * dy);
    };
    float R = dist(0, 2) + dist(1, 3) - dist(1, 2) - dist(0, 3);

    const float* g = ge + (rb << 6);
    float q = 0.f;
#pragma unroll 4
    for (int j = 0; j < HD; ++j) {
        float f = g[j] + bL12h[j];
#pragma unroll
        for (int k = 0; k < 8; ++k) f += Y[k] * WL12h[j * 8 + k];
        f = f > 0.f ? f : 0.f;
        q += f * Wt5[j];
    }
    q += R * Wt5[HD] + bt5[0];

    R_out[row] = R;
    Q_out[row] = q;
}

// ---------------------------------------------------------------------------
// Host-side launch
// ---------------------------------------------------------------------------
extern "C" void kernel_launch(void* const* d_in, const int* in_sizes, int n_in,
                              void* d_out, int out_size, void* d_ws, size_t ws_size,
                              hipStream_t stream) {
    const float* adj     = (const float*)d_in[0];
    const float* x       = (const float*)d_in[1];
    const int*   label   = (const int*)d_in[2];
    const int*   actions = (const int*)d_in[3];
    const float* W1   = (const float*)d_in[4];
    const float* b1   = (const float*)d_in[5];
    const float* W2   = (const float*)d_in[6];
    const float* b2   = (const float*)d_in[7];
    const float* Wih0 = (const float*)d_in[8];
    const float* bih0 = (const float*)d_in[9];
    const float* Whh0 = (const float*)d_in[10];
    const float* bhh0 = (const float*)d_in[11];
    const float* Wih1 = (const float*)d_in[12];
    const float* bih1 = (const float*)d_in[13];
    const float* Whh1 = (const float*)d_in[14];
    const float* bhh1 = (const float*)d_in[15];
    const float* Wt6  = (const float*)d_in[16];
    const float* bt6  = (const float*)d_in[17];
    const float* WL12h = (const float*)d_in[18];
    const float* bL12h = (const float*)d_in[19];
    const float* Wt5   = (const float*)d_in[20];
    const float* bt5   = (const float*)d_in[21];

    // output: (R[16384], h[32768*64], Q_sa[16384]) concatenated, f32
    float* out   = (float*)d_out;
    float* R_out = out;
    float* h_out = out + BNA;
    float* Q_out = out + BNA + BN * HD;

    // workspace layout (~28 MB)
    float*          xl1  = (float*)d_ws;                      // BN*HD f32
    float*          hws  = xl1 + BN * HD;                     // BN*HD f32
    unsigned short* hbf  = (unsigned short*)(hws + BN * HD);  // BN*HD bf16 row-major
    unsigned short* hbfT = hbf + BN * HD;                     // BN*HD bf16 transposed
    unsigned short* n1bf = hbfT + BN * HD;                    // BN*HD bf16
    float*          ge   = (float*)(n1bf + BN * HD);          // B*HD f32

    // GNN step 1 (h==0) fused with xl1
    init_kernel<<<(BN * HD) / 256, 256, 0, stream>>>(x, W1, b1, b2, xl1, hws,
                                                     hbf, hbfT);
    // GNN step 2
    gnn_adj_kernel<<<BB * 16, 256, 0, stream>>>(adj, hbfT, n1bf);
    gnn_fc_kernel<<<BN / 128, 256, 0, stream>>>(n1bf, W2, b2, xl1, hws, hbf, hbfT);
    // GNN step 3 (adj re-read hits the 192MB L2); f32 h goes straight to d_out
    gnn_adj_kernel<<<BB * 16, 256, 0, stream>>>(adj, hbfT, n1bf);
    gnn_fc_kernel<<<BN / 128, 256, 0, stream>>>(n1bf, W2, b2, xl1, h_out, hbf, hbfT);
    // sequential RNN scan + ge
    rnn_kernel<<<1, 256, 0, stream>>>(hbf, Wih0, bih0, Whh0, bhh0,
                                      Wih1, bih1, Whh1, bhh1, Wt6, bt6, ge);
    // R + Q_sa
    action_kernel<<<BNA / 256, 256, 0, stream>>>(label, actions, x, ge,
                                                 WL12h, bL12h, Wt5, bt5,
                                                 R_out, Q_out);
}
